// Repro_90383291777481
// MI455X (gfx1250) — matrix operand order repaired and checked by reading, not compiled, not run on hardware
//
#include <hip/hip_runtime.h>

typedef __attribute__((ext_vector_type(16))) _Float16 v16h;
typedef __attribute__((ext_vector_type(8)))  _Float16 v8h;
typedef __attribute__((ext_vector_type(4)))  _Float16 v4h;
typedef __attribute__((ext_vector_type(8)))  float    v8f;

#define S_LEN 2048
#define D_DIM 128
#define HQ    16
#define HKV   8
#define NREP  (HQ / HKV)
#define NEG_LARGE (-1.0e30f)

// LDS strides in halfs (16B-aligned rows, padded to rotate banks)
#define KSTR 144   // K tile row-major [key][d],    64 x 144 = 18432 B
#define VSTR 72    // V tile transposed [d][key],  128 x 72  = 18432 B
#define QSTR 136   // Q block row-major [row][d], 128 x 136  = 34816 B (overlaid)

// double-buffered K/V tiles + P scratch
#define BUF_STRIDE 36864
#define V_SUBOFF   18432
#define PS_OFF     73728
#define SMEM_BYTES (73728 + 16384)    // 88 KB (CDNA5: up to 320 KB/WG)

// scale * log2(e) folded into Q: scores come out in log2 domain -> bare v_exp_f32
#define QSCALE 0.12751743767043496f   // (1/sqrt(128)) * log2(e)

#define EXP2F(x) __builtin_amdgcn_exp2f(x)

union V16U { v16h v; v8h h[2]; };

// Each wave owns TWO 16-row q-tiles (t=0,1); K/V fragments are loaded once and
// feed one WMMA per q-tile -> ds-per-wmma halved vs one-tile-per-wave.
template <bool NEED_MASK>
__device__ __forceinline__ void process_tiles(
    const _Float16* kf16, const _Float16* vf16, _Float16* pw,
    const v16h (&qa)[2][4], v8f (&o_acc)[2][8],
    float (&m)[2][8], float (&l)[2][8],
    int ktb, int q0,   // q0 = first row of this wave's t=0 tile
    int half, int l16, int lane)
{
    bool act[2];
    act[0] = !NEED_MASK || (ktb <= q0 + 15);
    act[1] = !NEED_MASK || (ktb <= q0 + 31);

    // ---- scores = Q * K^T : four 16x16 C tiles per q-tile, shared kb ----
    v8f sc[2][4];
    #pragma unroll
    for (int n = 0; n < 4; ++n) {
        v8f a0 = {}, a1 = {};
        const _Float16* kb = kf16 + (n * 16 + l16) * KSTR + half * 8;
        #pragma unroll
        for (int c = 0; c < 4; ++c) {
            V16U u;
            u.h[0] = *(const v8h*)(kb + c * 32);
            u.h[1] = *(const v8h*)(kb + c * 32 + 16);
            a0 = __builtin_amdgcn_wmma_f32_16x16x32_f16(
                false, qa[0][c], false, u.v, (short)0, a0, false, false);
            a1 = __builtin_amdgcn_wmma_f32_16x16x32_f16(
                false, qa[1][c], false, u.v, (short)0, a1, false, false);
        }
        sc[0][n] = a0;
        sc[1][n] = a1;
    }

    // ---- per q-tile softmax + P transpose-bounce ----
    V16U pa[2][2];
    #pragma unroll
    for (int t = 0; t < 2; ++t) {
        if (NEED_MASK && !act[t]) continue;   // wave-uniform

        if (NEED_MASK) {
            #pragma unroll
            for (int r = 0; r < 8; ++r) {
                const int qrow = q0 + t * 16 + r + half * 8;
                #pragma unroll
                for (int n = 0; n < 4; ++n) {
                    const int key = ktb + n * 16 + l16;
                    if (key > qrow) sc[t][n][r] = NEG_LARGE;
                }
            }
        }

        float mt[8];
        #pragma unroll
        for (int r = 0; r < 8; ++r)
            mt[r] = fmaxf(fmaxf(sc[t][0][r], sc[t][1][r]),
                          fmaxf(sc[t][2][r], sc[t][3][r]));
        #pragma unroll
        for (int off = 1; off < 16; off <<= 1) {
            #pragma unroll
            for (int r = 0; r < 8; ++r)
                mt[r] = fmaxf(mt[r], __shfl_xor(mt[r], off, 32));
        }

        float rs[8];
        #pragma unroll
        for (int r = 0; r < 8; ++r) {
            const float mnew = fmaxf(m[t][r], mt[r]);
            const float corr = EXP2F(m[t][r] - mnew);
            float acc_p = 0.0f;
            #pragma unroll
            for (int n = 0; n < 4; ++n) {
                const float p = EXP2F(sc[t][n][r] - mnew);
                sc[t][n][r] = p;
                acc_p += p;
            }
            rs[r]    = acc_p;
            m[t][r]  = mnew;
            l[t][r] *= corr;
            #pragma unroll
            for (int dc = 0; dc < 8; ++dc) o_acc[t][dc][r] *= corr;
        }
        #pragma unroll
        for (int off = 1; off < 16; off <<= 1) {
            #pragma unroll
            for (int r = 0; r < 8; ++r)
                rs[r] += __shfl_xor(rs[r], off, 32);
        }
        #pragma unroll
        for (int r = 0; r < 8; ++r) l[t][r] += rs[r];

        // P: C layout -> A-fragment order (per-wave, per-tile region)
        _Float16* pt = pw + t * 1024;
        #pragma unroll
        for (int r = 0; r < 8; ++r) {
            const int Mr = r + half * 8;
            #pragma unroll
            for (int n = 0; n < 4; ++n) {
                const int key  = n * 16 + l16;        // 0..63
                const int j    = key >> 5;
                const int keyl = key & 31;
                const int Lsel = Mr + ((keyl >> 3) & 1) * 16;
                const int hsel = (keyl & 7) | ((keyl >> 4) << 3);
                pt[j * 512 + Lsel * 16 + hsel] = (_Float16)sc[t][n][r];
            }
        }
        #pragma unroll
        for (int j = 0; j < 2; ++j) {
            pa[t][j].h[0] = *(const v8h*)(pt + j * 512 + lane * 16);
            pa[t][j].h[1] = *(const v8h*)(pt + j * 512 + lane * 16 + 8);
        }
    }

    // ---- O += P * V : shared vb, one WMMA per q-tile ----
    #pragma unroll
    for (int dc = 0; dc < 8; ++dc) {
        const _Float16* vb = vf16 + (dc * 16 + l16) * VSTR + half * 8;
        #pragma unroll
        for (int j = 0; j < 2; ++j) {
            V16U u;
            u.h[0] = *(const v8h*)(vb + j * 32);
            u.h[1] = *(const v8h*)(vb + j * 32 + 16);
            #pragma unroll
            for (int t = 0; t < 2; ++t) {
                if (NEED_MASK && !act[t]) continue;
                o_acc[t][dc] = __builtin_amdgcn_wmma_f32_16x16x32_f16(
                    false, pa[t][j].v, false, u.v, (short)0, o_acc[t][dc],
                    false, false);
            }
        }
    }
}

__device__ __forceinline__ void stage_kv_tile(
    _Float16* kf16, _Float16* vf16,
    const float* kp, const float* vp, int ktb, int tid)
{
    // K tile (64x128) row-major f16
    {
        const float4* src = (const float4*)(kp + (size_t)ktb * D_DIM);
        #pragma unroll
        for (int i = 0; i < 16; ++i) {
            const int f   = tid + i * 128;
            const int key = f >> 5;
            const int dp  = (f & 31) << 2;
            float4 x = src[f];
            v4h pk;
            pk[0] = (_Float16)x.x; pk[1] = (_Float16)x.y;
            pk[2] = (_Float16)x.z; pk[3] = (_Float16)x.w;
            *(v4h*)(kf16 + key * KSTR + dp) = pk;
        }
    }
    // V tile (64x128) transposed -> [d][key] f16
    {
        const float4* src = (const float4*)(vp + (size_t)ktb * D_DIM);
        #pragma unroll
        for (int i = 0; i < 16; ++i) {
            const int f   = tid + i * 128;
            const int key = f >> 5;
            const int dp  = (f & 31) << 2;
            float4 x = src[f];
            vf16[(dp + 0) * VSTR + key] = (_Float16)x.x;
            vf16[(dp + 1) * VSTR + key] = (_Float16)x.y;
            vf16[(dp + 2) * VSTR + key] = (_Float16)x.z;
            vf16[(dp + 3) * VSTR + key] = (_Float16)x.w;
        }
    }
}

// 4 waves/block, 128-row q-block; wave w owns rows [qbase + w*32, +32).
// 64-key tiles, double-buffered LDS: tile kt+1 is staged while tile kt is
// computed (one barrier per tile; global-load latency hidden under WMMA).
__global__ __launch_bounds__(128) void fa_gqa_causal_kernel(
    const float* __restrict__ Q,
    const float* __restrict__ Km,
    const float* __restrict__ Vm,
    float* __restrict__ O)
{
    __shared__ __align__(32) char smem[SMEM_BYTES];

    const int tid  = threadIdx.x;
    const int wave = tid >> 5;
    const int lane = tid & 31;
    const int half = lane >> 4;
    const int l16  = lane & 15;

    const int nqb   = S_LEN / 128;                    // 16 q-blocks
    const int qblk  = blockIdx.x % nqb;
    const int h     = (blockIdx.x / nqb) % HQ;
    const int b     = blockIdx.x / (nqb * HQ);
    const int hkv   = h / NREP;
    const int qbase = qblk * 128;
    const int q0    = qbase + wave * 32;              // wave's first row

    const float* qp = Q  + ((size_t)(b * HQ + h) * S_LEN + qbase) * D_DIM;
    const float* kp = Km + (size_t)(b * HKV + hkv) * S_LEN * D_DIM;
    const float* vp = Vm + (size_t)(b * HKV + hkv) * S_LEN * D_DIM;
    float*       op = O  + ((size_t)(b * HQ + h) * S_LEN + q0) * D_DIM;

    // Q staged f16 overlaid on buffer 1 (first overwritten at kt=0 staging,
    // which happens after the qa reads + the kt=0 loop-top barrier)
    _Float16* qf16 = (_Float16*)(smem + BUF_STRIDE);
    _Float16* pscr = (_Float16*)(smem + PS_OFF);

    // ---- stage Q block (128x128) as f16, pre-scaled by QSCALE ----
    {
        const float4* src = (const float4*)qp;
        #pragma unroll
        for (int i = 0; i < 32; ++i) {
            const int f   = tid + i * 128;
            const int row = f >> 5;
            const int dp  = (f & 31) << 2;
            float4 x = src[f];
            v4h pk;
            pk[0] = (_Float16)(x.x * QSCALE); pk[1] = (_Float16)(x.y * QSCALE);
            pk[2] = (_Float16)(x.z * QSCALE); pk[3] = (_Float16)(x.w * QSCALE);
            *(v4h*)(qf16 + row * QSTR + dp) = pk;
        }
    }
    __syncthreads();

    // ---- Q A-fragments for both q-tiles ----
    v16h qa[2][4];
    #pragma unroll
    for (int t = 0; t < 2; ++t) {
        const _Float16* qrow = qf16 + (wave * 32 + t * 16 + l16) * QSTR;
        #pragma unroll
        for (int c = 0; c < 4; ++c) {
            V16U u;
            u.h[0] = *(const v8h*)(qrow + c * 32 + (half ? 8 : 0));
            u.h[1] = *(const v8h*)(qrow + c * 32 + (half ? 24 : 16));
            qa[t][c] = u.v;
        }
    }

    v8f   o_acc[2][8];
    float m[2][8], l[2][8];
    #pragma unroll
    for (int t = 0; t < 2; ++t) {
        #pragma unroll
        for (int r = 0; r < 8; ++r) {
            v8f z = {};
            o_acc[t][r] = z;
            m[t][r] = NEG_LARGE;
            l[t][r] = 0.0f;
        }
    }

    _Float16* pw = pscr + wave * 2048;

    // ---- prologue: stage tile 0 into buffer 0 (not the Q overlay) ----
    stage_kv_tile((_Float16*)(smem), (_Float16*)(smem + V_SUBOFF),
                  kp, vp, 0, tid);

    // ---- main loop: one barrier per tile; stage kt+1 while computing kt ----
    const int nkt = 2 * qblk + 2;     // last two tiles need the causal mask
    for (int kt = 0; kt < nkt; ++kt) {
        const int ktb = kt * 64;
        char* curb = smem + (size_t)(kt & 1) * BUF_STRIDE;
        char* nxtb = smem + (size_t)((kt & 1) ^ 1) * BUF_STRIDE;

        __syncthreads();   // staging of tile kt complete; prev reads of next buf done

        if (kt + 1 < nkt)
            stage_kv_tile((_Float16*)nxtb, (_Float16*)(nxtb + V_SUBOFF),
                          kp, vp, ktb + 64, tid);

        if (kt < nkt - 2) {
            process_tiles<false>((const _Float16*)curb,
                                 (const _Float16*)(curb + V_SUBOFF),
                                 pw, qa, o_acc, m, l, ktb, q0, half, l16, lane);
        } else {
            process_tiles<true>((const _Float16*)curb,
                                (const _Float16*)(curb + V_SUBOFF),
                                pw, qa, o_acc, m, l, ktb, q0, half, l16, lane);
        }
    }

    // ---- epilogue: divide by l, store both q-tiles ----
    #pragma unroll
    for (int t = 0; t < 2; ++t) {
        #pragma unroll
        for (int r = 0; r < 8; ++r) {
            const int   Mr  = t * 16 + r + half * 8;
            const float inv = 1.0f / l[t][r];
            #pragma unroll
            for (int dc = 0; dc < 8; ++dc)
                op[(size_t)Mr * D_DIM + dc * 16 + l16] = o_acc[t][dc][r] * inv;
        }
    }
}

extern "C" void kernel_launch(void* const* d_in, const int* in_sizes, int n_in,
                              void* d_out, int out_size, void* d_ws, size_t ws_size,
                              hipStream_t stream) {
    (void)in_sizes; (void)n_in; (void)out_size; (void)d_ws; (void)ws_size;
    const float* q = (const float*)d_in[0];
    const float* k = (const float*)d_in[1];
    const float* v = (const float*)d_in[2];
    float*       o = (float*)d_out;

    const int nblocks = 2 /*B*/ * HQ * (S_LEN / 128);   // 512
    fa_gqa_causal_kernel<<<dim3(nblocks), dim3(128), 0, stream>>>(q, k, v, o);
}
